// GraphConvLayer_88038239634289
// MI455X (gfx1250) — matrix operand order repaired and checked by reading, not compiled, not run on hardware
//
#include <hip/hip_runtime.h>
#include <math.h>

// ---------------------------------------------------------------------------
// GraphConv layer for MI455X (gfx1250), wave32 + WMMA bf16.
//
// Factorization: messages = FFN_p(x[nbr]) * w  ==>  P = FFN_p(x) per node,
// then edge scatter agg[dst] += w * P[nbr]; update = FFN_u([x|agg]) + l2norm.
// ---------------------------------------------------------------------------

typedef __attribute__((ext_vector_type(16))) __bf16 v16bf;
typedef __attribute__((ext_vector_type(8)))  float  v8f;

__device__ __forceinline__ float gelu_exact(float x) {
    return 0.5f * x * (1.0f + erff(x * 0.70710678118654752f));
}

__device__ __forceinline__ v8f wmma_bf16(v16bf a, v16bf b, v8f c) {
    // D(16x16 f32) = A(16x32 bf16) x B(32x16 bf16) + C
    return __builtin_amdgcn_wmma_f32_16x16x32_bf16(
        /*neg_a=*/false, a, /*neg_b=*/false, b,
        /*c_mod=*/(short)0, c, /*reuse_a=*/false, /*reuse_b=*/false);
}

// A-fragment from fp32 global row; rowk points at (row, 32*kchunk).
// A layout (matches B): lane half h: elems 0..15 = K 16h..16h+15
__device__ __forceinline__ v16bf load_a_f32(const float* rowk, int half) {
    const float4* p0 = (const float4*)(rowk + 16 * half);
    const float4* p1 = (const float4*)(rowk + 16 * half + 8);
    float4 a0 = p0[0], a1 = p0[1], b0 = p1[0], b1 = p1[1];
    float t[16] = {a0.x, a0.y, a0.z, a0.w, a1.x, a1.y, a1.z, a1.w,
                   b0.x, b0.y, b0.z, b0.w, b1.x, b1.y, b1.z, b1.w};
    v16bf r;
#pragma unroll
    for (int j = 0; j < 16; ++j) r[j] = (__bf16)t[j];
    return r;
}

// A-fragment from bf16 LDS tile (row stride LDS_STR elems)
#define LDS_STR 68
__device__ __forceinline__ v16bf load_a_lds(const __bf16* t, int m, int kb, int half) {
    const __bf16* p = t + m * LDS_STR + kb + 16 * half;
    v16bf r;
#pragma unroll
    for (int j = 0; j < 8; ++j) { r[j] = p[j]; r[8 + j] = p[8 + j]; }
    return r;
}

// B-fragment: Wt is [Nout][K] bf16 (pre-transposed); one contiguous 32B load.
// B layout used here (matches A): lane col n = l&15, elems 0..15 = K 16h..16h+15
__device__ __forceinline__ v16bf load_b(const __bf16* Wt, int col, int Kt, int kb, int half) {
    return *(const v16bf*)(Wt + (size_t)col * Kt + kb + 16 * half);
}

// -------------------------- small utility kernels --------------------------

__global__ void transpose_to_bf16(const float* __restrict__ src, __bf16* __restrict__ dst,
                                  int K, int Nn) {
    int i = blockIdx.x * blockDim.x + threadIdx.x;
    if (i < K * Nn) {
        int k = i / Nn, n = i % Nn;
        dst[(size_t)n * K + k] = (__bf16)src[(size_t)k * Nn + n];
    }
}

__global__ void zero_f32(float* __restrict__ p, int n) {
    int i = blockIdx.x * blockDim.x + threadIdx.x;
    if (i < n) p[i] = 0.0f;
}

// ---------------- per-node "prepare" FFN:  P = FFN_p(x)  -------------------
// one wave per 16-row tile; K=64 (2 K-chunks), Nout=64 (4 N-tiles)

__global__ __launch_bounds__(128) void node_ffn_prep(
    const float* __restrict__ x,
    const __bf16* __restrict__ Wt1, const float* __restrict__ b1,
    const __bf16* __restrict__ Wt2, const float* __restrict__ b2,
    float* __restrict__ P, int ntiles) {
    __shared__ __bf16 lds[4][16 * LDS_STR];
    const int wave = threadIdx.x >> 5;
    const int lane = threadIdx.x & 31;
    const int tile = blockIdx.x * 4 + wave;
    const bool active = tile < ntiles;
    const int m = lane & 15, half = lane >> 4, n = lane & 15;
    __bf16* t = lds[wave];
    v8f c1[4];

    if (active) {
        const float* xrow = x + (size_t)(tile * 16 + m) * 64;
        v16bf a0 = load_a_f32(xrow, half);
        v16bf a1 = load_a_f32(xrow + 32, half);
#pragma unroll
        for (int nt = 0; nt < 4; ++nt) {
            float bb = b1[nt * 16 + n];
            v8f acc;
#pragma unroll
            for (int r = 0; r < 8; ++r) acc[r] = bb;
            acc = wmma_bf16(a0, load_b(Wt1, nt * 16 + n, 64, 0, half), acc);
            acc = wmma_bf16(a1, load_b(Wt1, nt * 16 + n, 64, 32, half), acc);
            c1[nt] = acc;
        }
        // GELU and stage transposed copy into LDS as bf16 (C layout -> [m][k])
#pragma unroll
        for (int nt = 0; nt < 4; ++nt)
#pragma unroll
            for (int r = 0; r < 8; ++r) {
                float g = gelu_exact(c1[nt][r]);
                t[(r + 8 * half) * LDS_STR + nt * 16 + n] = (__bf16)g;
            }
    }
    __syncthreads();
    if (active) {
        v16bf a0 = load_a_lds(t, m, 0, half);
        v16bf a1 = load_a_lds(t, m, 32, half);
#pragma unroll
        for (int nt = 0; nt < 4; ++nt) {
            float bb = b2[nt * 16 + n];
            v8f acc;
#pragma unroll
            for (int r = 0; r < 8; ++r) acc[r] = bb;
            acc = wmma_bf16(a0, load_b(Wt2, nt * 16 + n, 64, 0, half), acc);
            acc = wmma_bf16(a1, load_b(Wt2, nt * 16 + n, 64, 32, half), acc);
#pragma unroll
            for (int r = 0; r < 8; ++r) {
                int row = tile * 16 + r + 8 * half;
                P[(size_t)row * 64 + nt * 16 + n] = gelu_exact(acc[r]);
            }
        }
    }
}

// ---------------- edge scatter: agg[dst] += w * P[nbr] ---------------------
// one wave per edge; P & agg are L2-resident (12.8 MB each)

__global__ __launch_bounds__(256) void edge_scatter(
    const int* __restrict__ edges, const float* __restrict__ ew,
    const float* __restrict__ P, float* __restrict__ agg, int E) {
    const int lane = threadIdx.x & 31;
    const int gw = (blockIdx.x * blockDim.x + threadIdx.x) >> 5;
    const int nw = (gridDim.x * blockDim.x) >> 5;
    for (int e = gw; e < E; e += nw) {
        int dst = edges[e];            // node_indices (segment ids)
        int src = edges[E + e];        // neighbour_indices
        float w = ew[e];
        if (e + nw < E) {              // prefetch next edge record of this wave
            __builtin_prefetch(&edges[e + nw], 0, 0);
            __builtin_prefetch(&edges[E + e + nw], 0, 0);
        }
        const float* p = P + (size_t)src * 64;
        float* a = agg + (size_t)dst * 64;
        unsafeAtomicAdd(a + lane,      p[lane]      * w);
        unsafeAtomicAdd(a + 32 + lane, p[32 + lane] * w);
    }
}

// ---------------- update FFN + l2norm: out = norm(FFN_u([x|agg])) ----------
// K=128 (4 K-chunks: 2 from x, 2 from agg), Nout=64

__global__ __launch_bounds__(128) void node_update(
    const float* __restrict__ x, const float* __restrict__ agg,
    const __bf16* __restrict__ Wt1, const float* __restrict__ b1,   // [64][128]
    const __bf16* __restrict__ Wt2, const float* __restrict__ b2,   // [64][64]
    float* __restrict__ out, int ntiles) {
    __shared__ __bf16 lds[4][16 * LDS_STR];
    const int wave = threadIdx.x >> 5;
    const int lane = threadIdx.x & 31;
    const int tile = blockIdx.x * 4 + wave;
    const bool active = tile < ntiles;
    const int m = lane & 15, half = lane >> 4, n = lane & 15;
    __bf16* t = lds[wave];

    if (active) {
        const float* xrow = x + (size_t)(tile * 16 + m) * 64;
        const float* arow = agg + (size_t)(tile * 16 + m) * 64;
        v16bf a[4];
        a[0] = load_a_f32(xrow, half);
        a[1] = load_a_f32(xrow + 32, half);
        a[2] = load_a_f32(arow, half);
        a[3] = load_a_f32(arow + 32, half);
#pragma unroll
        for (int nt = 0; nt < 4; ++nt) {
            float bb = b1[nt * 16 + n];
            v8f acc;
#pragma unroll
            for (int r = 0; r < 8; ++r) acc[r] = bb;
#pragma unroll
            for (int c = 0; c < 4; ++c)
                acc = wmma_bf16(a[c], load_b(Wt1, nt * 16 + n, 128, 32 * c, half), acc);
#pragma unroll
            for (int r = 0; r < 8; ++r)
                t[(r + 8 * half) * LDS_STR + nt * 16 + n] = (__bf16)gelu_exact(acc[r]);
        }
    }
    __syncthreads();
    if (active) {
        v16bf a0 = load_a_lds(t, m, 0, half);
        v16bf a1 = load_a_lds(t, m, 32, half);
        v8f c2[4];
#pragma unroll
        for (int nt = 0; nt < 4; ++nt) {
            float bb = b2[nt * 16 + n];
            v8f acc;
#pragma unroll
            for (int r = 0; r < 8; ++r) acc[r] = bb;
            acc = wmma_bf16(a0, load_b(Wt2, nt * 16 + n, 64, 0, half), acc);
            acc = wmma_bf16(a1, load_b(Wt2, nt * 16 + n, 64, 32, half), acc);
#pragma unroll
            for (int r = 0; r < 8; ++r) acc[r] = gelu_exact(acc[r]);
            c2[nt] = acc;
        }
        // l2 normalize: row (r + 8*half) lives across the 16 lanes of this half
        float ss[8];
#pragma unroll
        for (int r = 0; r < 8; ++r) {
            float s = 0.0f;
#pragma unroll
            for (int nt = 0; nt < 4; ++nt) s += c2[nt][r] * c2[nt][r];
            ss[r] = s;
        }
#pragma unroll
        for (int mask = 1; mask < 16; mask <<= 1)
#pragma unroll
            for (int r = 0; r < 8; ++r) ss[r] += __shfl_xor(ss[r], mask, 32);
#pragma unroll
        for (int r = 0; r < 8; ++r) ss[r] = rsqrtf(fmaxf(ss[r], 1e-12f));
#pragma unroll
        for (int nt = 0; nt < 4; ++nt)
#pragma unroll
            for (int r = 0; r < 8; ++r) {
                int row = tile * 16 + r + 8 * half;
                out[(size_t)row * 64 + nt * 16 + n] = c2[nt][r] * ss[r];
            }
    }
}

// ------------------------------- launcher ----------------------------------

extern "C" void kernel_launch(void* const* d_in, const int* in_sizes, int n_in,
                              void* d_out, int out_size, void* d_ws, size_t ws_size,
                              hipStream_t stream) {
    const float* x   = (const float*)d_in[0];
    const int*   edg = (const int*)d_in[1];
    const float* ew  = (const float*)d_in[2];
    const float* W1p = (const float*)d_in[3];
    const float* b1p = (const float*)d_in[4];
    const float* W2p = (const float*)d_in[5];
    const float* b2p = (const float*)d_in[6];
    const float* W1u = (const float*)d_in[7];
    const float* b1u = (const float*)d_in[8];
    const float* W2u = (const float*)d_in[9];
    const float* b2u = (const float*)d_in[10];
    float* out = (float*)d_out;

    const int N = in_sizes[0] / 64;
    const int E = in_sizes[2];
    const int ntiles = N / 16;

    // workspace layout (all offsets 256B-aligned)
    char* ws = (char*)d_ws;
    __bf16* Wt1p = (__bf16*)(ws + 0);              //  64x64  bf16 :  8 KB
    __bf16* Wt2p = (__bf16*)(ws + 8192);           //  64x64  bf16 :  8 KB
    __bf16* Wt1u = (__bf16*)(ws + 16384);          //  64x128 bf16 : 16 KB
    __bf16* Wt2u = (__bf16*)(ws + 32768);          //  64x64  bf16 :  8 KB
    float*  P    = (float*)(ws + 40960);           //  N x 64 f32
    float*  agg  = (float*)(ws + 40960 + (size_t)N * 64 * 4);

    // 1) weight prep (transpose to [n][k] bf16)
    transpose_to_bf16<<<(64 * 64 + 255) / 256, 256, 0, stream>>>(W1p, Wt1p, 64, 64);
    transpose_to_bf16<<<(64 * 64 + 255) / 256, 256, 0, stream>>>(W2p, Wt2p, 64, 64);
    transpose_to_bf16<<<(128 * 64 + 255) / 256, 256, 0, stream>>>(W1u, Wt1u, 128, 64);
    transpose_to_bf16<<<(64 * 64 + 255) / 256, 256, 0, stream>>>(W2u, Wt2u, 64, 64);

    // 2) zero the aggregation buffer
    zero_f32<<<(N * 64 + 255) / 256, 256, 0, stream>>>(agg, N * 64);

    // 3) per-node prepare-FFN (factored out of the edge loop)
    node_ffn_prep<<<(ntiles + 3) / 4, 128, 0, stream>>>(x, Wt1p, b1p, Wt2p, b2p, P, ntiles);

    // 4) edge gather/scale/scatter-add (L2-resident P and agg)
    edge_scatter<<<2048, 256, 0, stream>>>(edg, ew, P, agg, E);

    // 5) update FFN + l2 normalize
    node_update<<<(ntiles + 3) / 4, 128, 0, stream>>>(x, agg, Wt1u, b1u, Wt2u, b2u, out, ntiles);
}
